// d3GraphConvat_188978561285
// MI455X (gfx1250) — matrix operand order repaired and checked by reading, not compiled, not run on hardware
//
#include <hip/hip_runtime.h>

// ---------------------------------------------------------------------------
// GAT (dense) + temporal cosine mixing for MI455X (gfx1250, wave32, WMMA).
// Memory-bound on adj (134 MB, read twice ~ 11.5 us at 23.3 TB/s).
// Flash-style 2-pass softmax; bf16 WMMA GEMMs; adj streamed into LDS with
// double-buffered GLOBAL_LOAD_ASYNC_TO_LDS_B128 (ASYNCcnt) in pass 2.
// ---------------------------------------------------------------------------

#define T_DIM 8
#define N_DIM 2048
#define FIN   128
#define C_DIM 64
#define NEG_SLOPE 0.2f
#define COS_EPS 1e-8f
#define NJT (N_DIM / 32)

typedef __attribute__((ext_vector_type(16))) __bf16 v16bf;
typedef __attribute__((ext_vector_type(8)))  float  v8f;

// --- swizzle W (128x64 f32, row-major) into WMMA B-fragment tile layout ----
// tile (kb, ct): lane n, element i  <-  W[kb*32 + i + (i>=8?8:0) + 8*(n/16)][ct*16 + n%16]
__global__ void k_swizzle_W(const float* __restrict__ W, __bf16* __restrict__ Wb) {
  const int lane = threadIdx.x & 31;
  const int wid  = threadIdx.x >> 5;       // 16 waves, 1 block
  const int ct = wid & 3, kb = wid >> 2;
  const int n = lane & 15, half = lane >> 4;
  __bf16* dst = Wb + ((size_t)wid * 32 + lane) * 16;
#pragma unroll
  for (int i = 0; i < 16; ++i)
    dst[i] = (__bf16)W[(kb * 32 + i + (i >= 8 ? 8 : 0) + 8 * half) * C_DIM + ct * 16 + n];
}

// --- h = x @ W  via v_wmma_f32_16x16x32_bf16 (one 16x64 tile per wave) -----
__global__ void k_gemm_h(const float* __restrict__ x, const __bf16* __restrict__ Wb,
                         float* __restrict__ h) {
  const int lane = threadIdx.x & 31;
  const int wid  = blockIdx.x * 8 + (threadIdx.x >> 5);   // 256 thr = 8 waves
  const int mt = wid % (N_DIM / 16);
  const int t  = wid / (N_DIM / 16);
  const int m = lane & 15, half = lane >> 4;
  const int M0 = mt * 16;
  const float* xrow = x + ((size_t)t * N_DIM + M0 + m) * FIN;
  v8f acc[4] = {v8f{}, v8f{}, v8f{}, v8f{}};
#pragma unroll
  for (int kb = 0; kb < FIN / 32; ++kb) {
    v16bf afrag;
#pragma unroll
    for (int p = 0; p < 8; ++p) {                 // element pairs (2p, 2p+1)
      const int k = kb * 32 + 2 * p + (p >= 4 ? 8 : 0) + 8 * half;
      const float2 v = *(const float2*)(xrow + k);
      afrag[2 * p]     = (__bf16)v.x;
      afrag[2 * p + 1] = (__bf16)v.y;
    }
#pragma unroll
    for (int ct = 0; ct < 4; ++ct) {
      const v16bf bfrag = *(const v16bf*)(Wb + (((size_t)kb * 4 + ct) * 32 + lane) * 16);
      acc[ct] = __builtin_amdgcn_wmma_f32_16x16x32_bf16(
          false, afrag, false, bfrag, (short)0, acc[ct], false, false);
    }
  }
#pragma unroll
  for (int ct = 0; ct < 4; ++ct)
#pragma unroll
    for (int r = 0; r < 8; ++r)
      h[((size_t)t * N_DIM + M0 + r + 8 * half) * C_DIM + ct * 16 + m] = acc[ct][r];
}

// --- per-node attention coefficients: e_s = h·a_src, e_d = h·a_dst --------
__global__ void k_attn_coef(const float* __restrict__ h,
                            const float* __restrict__ a_src,
                            const float* __restrict__ a_dst,
                            float* __restrict__ es, float* __restrict__ ed) {
  const int idx = blockIdx.x * blockDim.x + threadIdx.x;   // t*N + n
  const float* hr = h + (size_t)idx * C_DIM;
  float s = 0.f, d = 0.f;
#pragma unroll 4
  for (int c = 0; c < C_DIM; ++c) { const float v = hr[c]; s += v * a_src[c]; d += v * a_dst[c]; }
  es[idx] = s; ed[idx] = d;
}

// --- swizzle h into bf16 WMMA B-fragment tiles (per-lane contiguous 32B) ---
__global__ void k_swizzle_h(const float* __restrict__ h, __bf16* __restrict__ hB) {
  const int lane = threadIdx.x & 31;
  const int wid  = blockIdx.x * 8 + (threadIdx.x >> 5);    // over T*(N/32)*4
  const int ct = wid & 3;
  const int jt = (wid >> 2) % NJT;
  const int t  = (wid >> 2) / NJT;
  const int n = lane & 15, half = lane >> 4;
  __bf16* dst = hB + ((size_t)wid * 32 + lane) * 16;
#pragma unroll
  for (int i = 0; i < 16; ++i)
    dst[i] = (__bf16)h[((size_t)t * N_DIM + jt * 32 + i + (i >= 8 ? 8 : 0) + 8 * half) * C_DIM + ct * 16 + n];
}

// --- pass 1: masked online-softmax stats per destination row ---------------
// mask[t,i,j] = adj[t,j,i] != 0 || i==j   (note the transpose in reference)
__global__ void k_softmax_stats(const int* __restrict__ adj,
                                const float* __restrict__ es,
                                const float* __restrict__ ed,
                                float* __restrict__ rmax, float* __restrict__ rinv) {
  const int idx = blockIdx.x * blockDim.x + threadIdx.x;   // t*N + i
  const int t = idx >> 11, i = idx & (N_DIM - 1);
  const int* adjcol = adj + (size_t)t * N_DIM * N_DIM + i; // stride N over j -> coalesced in i
  const float edv = ed[idx];
  const float* esr = es + t * N_DIM;
  float m = -3.0e38f, s = 0.f;
#pragma unroll 4
  for (int j = 0; j < N_DIM; ++j) {
    const int a = adjcol[(size_t)j * N_DIM];
    if (a != 0 || j == i) {
      float l = edv + esr[j];
      l = (l > 0.f) ? l : NEG_SLOPE * l;
      if (l <= m) s += __expf(l - m);
      else       { s = s * __expf(m - l) + 1.f; m = l; }
    }
  }
  rmax[idx] = m;
  rinv[idx] = 1.f / s;
}

// --- pass 2: out = softmax(logits) @ h + bias ------------------------------
// alpha recomputed on the fly; adj tiles streamed into LDS with async copies.
__global__ void __launch_bounds__(256)
k_aggregate(const int* __restrict__ adj,
            const float* __restrict__ es,
            const float* __restrict__ ed,
            const float* __restrict__ rmax,
            const float* __restrict__ rinv,
            const __bf16* __restrict__ hB,
            const float* __restrict__ bias,
            float* __restrict__ gout) {
  // per-wave double buffer: 2 x (32 rows x 16 cols) adj ints = 2 x 2KB
  __shared__ int adjbuf[8][2][512];
  const int lane = threadIdx.x & 31;
  const int w    = threadIdx.x >> 5;
  const int wid  = blockIdx.x * 8 + w;
  const int it = wid % (N_DIM / 16);
  const int t  = wid / (N_DIM / 16);
  const int m = lane & 15, half = lane >> 4;
  const int I0 = it * 16;
  const int gi = I0 + m;                       // this lane's destination row
  const float edv  = ed[t * N_DIM + gi];
  const float mmax = rmax[t * N_DIM + gi];
  const float sinv = rinv[t * N_DIM + gi];
  const float* esr = es + t * N_DIM;

  // async-copy addressing: chunk f = q*32+lane covers 16B each;
  // row = f>>2 within the 32-row j-tile, 4-int column group = f&3.
  const unsigned ldsBase = (unsigned)(size_t)(&adjbuf[w][0][0]);
  const int*     adjTile = adj + (size_t)t * N_DIM * N_DIM + I0;
  const size_t jtStride  = (size_t)32 * N_DIM;             // ints per j-tile

  auto prefetch = [&](int jt, int buf) {
#pragma unroll
    for (int q = 0; q < 4; ++q) {
      const int f = q * 32 + lane;
      const int row = f >> 2, chunk = f & 3;
      const long long ga = (long long)(size_t)(adjTile + (size_t)jt * jtStride +
                                               (size_t)row * N_DIM + chunk * 4);
      const unsigned lo = ldsBase + (unsigned)(buf * 2048 + f * 16);
      asm volatile("global_load_async_to_lds_b128 %0, %1, off"
                   :: "v"(lo), "v"(ga) : "memory");
    }
  };

  prefetch(0, 0);
  v8f acc[4] = {v8f{}, v8f{}, v8f{}, v8f{}};
  for (int jt = 0; jt < NJT; ++jt) {
    const int buf = jt & 1;
    if (jt + 1 < NJT) {
      prefetch(jt + 1, buf ^ 1);
      asm volatile("s_wait_asynccnt 0x4" ::: "memory"); // current tile landed
    } else {
      asm volatile("s_wait_asynccnt 0x0" ::: "memory");
    }
    v16bf afrag;                                // alpha tile 16x32, A-layout
#pragma unroll
    for (int i = 0; i < 16; ++i) {
      const int jl = i + (i >= 8 ? 8 : 0) + 8 * half;     // 0..31 in tile
      const int j  = jt * 32 + jl;
      const int a  = adjbuf[w][buf][jl * 16 + m];
      float p = 0.f;
      if (a != 0 || j == gi) {
        float l = edv + esr[j];
        l = (l > 0.f) ? l : NEG_SLOPE * l;
        p = __expf(l - mmax) * sinv;
      }
      afrag[i] = (__bf16)p;
    }
    const __bf16* btile = hB + ((size_t)(t * NJT + jt) * 4) * 512;
#pragma unroll
    for (int ct = 0; ct < 4; ++ct) {
      const v16bf bfrag = *(const v16bf*)(btile + ((size_t)ct * 32 + lane) * 16);
      acc[ct] = __builtin_amdgcn_wmma_f32_16x16x32_bf16(
          false, afrag, false, bfrag, (short)0, acc[ct], false, false);
    }
  }
#pragma unroll
  for (int ct = 0; ct < 4; ++ct)
#pragma unroll
    for (int r = 0; r < 8; ++r) {
      const int col = ct * 16 + m;
      gout[((size_t)t * N_DIM + I0 + r + 8 * half) * C_DIM + col] = acc[ct][r] + bias[col];
    }
}

// --- temporal cosine similarity between consecutive frames -----------------
__global__ void k_tem(const float* __restrict__ gout, float* __restrict__ tem) {
  const int idx = blockIdx.x * blockDim.x + threadIdx.x;   // t in [0,7), n
  if (idx >= (T_DIM - 1) * N_DIM) return;
  const float* a = gout + (size_t)idx * C_DIM;             // out[t, n]
  const float* b = a + (size_t)N_DIM * C_DIM;              // out[t+1, n]
  float dot = 0.f, na = 0.f, nb = 0.f;
#pragma unroll 4
  for (int c = 0; c < C_DIM; ++c) { dot += a[c] * b[c]; na += a[c] * a[c]; nb += b[c] * b[c]; }
  na = sqrtf(na); nb = sqrtf(nb);
  na = na > COS_EPS ? na : COS_EPS;
  nb = nb > COS_EPS ? nb : COS_EPS;
  tem[idx] = dot / (na * nb);
}

// --- final: out + t2_term + t1_term ---------------------------------------
__global__ void k_final(const float* __restrict__ gout, const float* __restrict__ tem,
                        const float* __restrict__ t1w, const float* __restrict__ t2w,
                        float* __restrict__ out) {
  const int idx = blockIdx.x * blockDim.x + threadIdx.x;   // ((t*N + n)*64 + d)
  const int d  = idx & 63;
  const int tn = idx >> 6;
  const int t = tn >> 11, n = tn & (N_DIM - 1);
  float acc = gout[idx];
  if (t > 0) {
    const float tv = tem[(t - 1) * N_DIM + n];
    const float* prow = gout + ((size_t)(t - 1) * N_DIM + n) * C_DIM;
    float s = 0.f;
#pragma unroll 4
    for (int c = 0; c < C_DIM; ++c) s += prow[c] * t2w[c * C_DIM + d];
    acc += tv * s;
  }
  if (t < T_DIM - 1) {
    const float tv = tem[t * N_DIM + n];
    const float* nrow = gout + ((size_t)(t + 1) * N_DIM + n) * C_DIM;
    float s = 0.f;
#pragma unroll 4
    for (int c = 0; c < C_DIM; ++c) s += nrow[c] * t1w[c * C_DIM + d];
    acc += tv * s;
  }
  out[idx] = acc;
}

extern "C" void kernel_launch(void* const* d_in, const int* in_sizes, int n_in,
                              void* d_out, int out_size, void* d_ws, size_t ws_size,
                              hipStream_t stream) {
  const float* x     = (const float*)d_in[0];
  const int*   adj   = (const int*)d_in[1];
  const float* W     = (const float*)d_in[2];
  const float* a_src = (const float*)d_in[3];
  const float* a_dst = (const float*)d_in[4];
  const float* bias  = (const float*)d_in[5];
  const float* t1w   = (const float*)d_in[6];
  const float* t2w   = (const float*)d_in[7];
  float* out = (float*)d_out;

  // workspace carve-out (~10.6 MB total)
  char* p = (char*)d_ws;
  float*  h    = (float*)p;  p += (size_t)T_DIM * N_DIM * C_DIM * sizeof(float);
  __bf16* hB   = (__bf16*)p; p += (size_t)T_DIM * N_DIM * C_DIM * sizeof(__bf16);
  __bf16* Wb   = (__bf16*)p; p += (size_t)FIN * C_DIM * sizeof(__bf16);
  float*  es   = (float*)p;  p += (size_t)T_DIM * N_DIM * sizeof(float);
  float*  ed   = (float*)p;  p += (size_t)T_DIM * N_DIM * sizeof(float);
  float*  rmax = (float*)p;  p += (size_t)T_DIM * N_DIM * sizeof(float);
  float*  rinv = (float*)p;  p += (size_t)T_DIM * N_DIM * sizeof(float);
  float*  gout = (float*)p;  p += (size_t)T_DIM * N_DIM * C_DIM * sizeof(float);
  float*  tem  = (float*)p;  p += (size_t)(T_DIM - 1) * N_DIM * sizeof(float);
  (void)ws_size; (void)n_in; (void)in_sizes; (void)out_size;

  k_swizzle_W<<<1, 512, 0, stream>>>(W, Wb);
  k_gemm_h<<<(T_DIM * (N_DIM / 16)) / 8, 256, 0, stream>>>(x, Wb, h);
  k_attn_coef<<<(T_DIM * N_DIM) / 256, 256, 0, stream>>>(h, a_src, a_dst, es, ed);
  k_swizzle_h<<<(T_DIM * NJT * 4) / 8, 256, 0, stream>>>(h, hB);
  k_softmax_stats<<<(T_DIM * N_DIM) / 256, 256, 0, stream>>>(adj, es, ed, rmax, rinv);
  k_aggregate<<<(T_DIM * (N_DIM / 16)) / 8, 256, 0, stream>>>(adj, es, ed, rmax, rinv, hB, bias, gout);
  k_tem<<<((T_DIM - 1) * N_DIM + 255) / 256, 256, 0, stream>>>(gout, tem);
  k_final<<<(T_DIM * N_DIM * C_DIM) / 256, 256, 0, stream>>>(gout, tem, t1w, t2w, out);
}
